// DRQN_25348896981628
// MI455X (gfx1250) — matrix operand order repaired and checked by reading, not compiled, not run on hardware
//
#include <hip/hip_runtime.h>

// ---------------- types ----------------
typedef __attribute__((ext_vector_type(16))) _Float16 v16h;
typedef __attribute__((ext_vector_type(8)))  float    v8f;
typedef __attribute__((ext_vector_type(4)))  int      v4i_;

// ---------------- problem constants ----------------
constexpr int Lc   = 6;
constexpr int Bc   = 1024;
constexpr int Nc   = 512;
constexpr int Gc   = 6;       // one-hot width (only col 0 is set)
constexpr int Hc   = 1024;
constexpr int AHc  = 256;
constexpr int ACTc = 12972;

constexpr int Mc   = Lc * Bc;        // 6144 rows
constexpr int K1c  = 544;            // (N+G)=518 padded to 17*32
constexpr int N1c  = AHc + Hc;       // 1280 = concat attention + encoder layer-1
constexpr int ACTP = 13056;          // ACT padded to 102*128

// ---------------- CDNA5 async global->LDS support ----------------
#if defined(__gfx1250__) && __has_builtin(__builtin_amdgcn_global_load_async_to_lds_b128)
#define HAVE_ASYNC_LDS 1
typedef __attribute__((address_space(1))) v4i_* gvec_p;   // global int4*
typedef __attribute__((address_space(3))) v4i_* lvec_p;   // LDS int4*
#define ASYNC_B128(gp, lp) \
  __builtin_amdgcn_global_load_async_to_lds_b128((gvec_p)(gp), (lvec_p)(lp), 0, 0)
#if __has_builtin(__builtin_amdgcn_s_wait_asynccnt)
#define WAIT_ASYNC0() __builtin_amdgcn_s_wait_asynccnt(0)
#else
#define WAIT_ASYNC0() asm volatile("s_wait_asynccnt 0" ::: "memory")
#endif
#endif

// ==================================================================
// Prep kernels
// ==================================================================

// inpH[m][k] (k<512: x, k==512: 1.0 (pos one-hot l0=0), else 0)  [Mc x K1c]
__global__ __launch_bounds__(256) void k_pack_inp(const float* __restrict__ x,
                                                  _Float16* __restrict__ inpH) {
  size_t idx = (size_t)blockIdx.x * 256 + threadIdx.x;
  if (idx >= (size_t)Mc * K1c) return;
  int m = (int)(idx / K1c), k = (int)(idx % K1c);
  float v = (k < Nc) ? x[(size_t)m * Nc + k] : (k == Nc ? 1.0f : 0.0f);
  inpH[idx] = (_Float16)v;
}

// Tiled transpose+convert: wt[n][k] = w[k][n], zero-padded to [Np x Kp].
// 32x32 tile via LDS: coalesced 128B reads of w, coalesced 64B f16 writes.
__global__ __launch_bounds__(256) void k_transpose_t(const float* __restrict__ w,
                                                     _Float16* __restrict__ wt,
                                                     int K, int Kp, int N, int Np) {
  __shared__ float tile[32][33];
  const int kb = blockIdx.x * 32;   // k block
  const int nb = blockIdx.y * 32;   // n block
  const int tx = threadIdx.x & 31, ty = threadIdx.x >> 5;   // 32 x 8
#pragma unroll
  for (int i = ty; i < 32; i += 8) {
    int k = kb + i, n = nb + tx;
    tile[i][tx] = (k < K && n < N) ? w[(size_t)k * N + n] : 0.0f;
  }
  __syncthreads();
#pragma unroll
  for (int i = ty; i < 32; i += 8) {
    int n = nb + i, k = kb + tx;
    if (n < Np && k < Kp) wt[(size_t)n * Kp + k] = (_Float16)tile[tx][i];
  }
}

// biasCat = ab1 ++ eb1  [N1c]
__global__ __launch_bounds__(256) void k_biascat(const float* __restrict__ ab1,
                                                 const float* __restrict__ eb1,
                                                 float* __restrict__ bc) {
  int i = blockIdx.x * 256 + threadIdx.x;
  if (i >= N1c) return;
  bc[i] = (i < AHc) ? ab1[i] : eb1[i - AHc];
}

// ==================================================================
// Tiled WMMA GEMM:  C[M x N] = op(A[M x K(lda)] * Bt[N x K(ldb)]^T + bias)
// BM=BN=128, BK=32, 256 threads = 8 waves (2x4), wave tile 64x32.
// Double-buffered LDS; async global->LDS copies when available.
// ==================================================================
template <bool RELU, bool SCALE_ATT, bool OUT_F16>
__global__ __launch_bounds__(256) void k_gemm(
    const _Float16* __restrict__ A, int lda,
    const _Float16* __restrict__ Bt, int ldb,
    const float* __restrict__ bias,
    const float* __restrict__ att,
    void* __restrict__ Cout, int ldc,
    int Nvalid, int K) {
  constexpr int BM = 128, BN = 128, BK = 32, PAD = 8, LDW = BK + PAD;
  __shared__ _Float16 As[2][BM][LDW];
  __shared__ _Float16 Bs[2][BN][LDW];

  const int tid   = threadIdx.x;
  const int wave  = tid >> 5;
  const int lane  = tid & 31;
  const int l15   = lane & 15;
  const int hi    = lane >> 4;      // 0 or 1
  const int waveM = wave >> 2;      // 0..1
  const int waveN = wave & 3;       // 0..3

  const int m0 = blockIdx.y * BM;
  const int n0 = blockIdx.x * BN;

  v8f acc[4][2] = {};
  const int nkt = K / BK;

#if defined(HAVE_ASYNC_LDS)
  // ---------------- async global->LDS double-buffered pipeline ----------------
  auto issue_copy = [&](int kt, int buf) {
#pragma unroll
    for (int i = 0; i < 2; ++i) {
      int c   = tid + i * 256;          // chunk id in [0,512)
      int row = c >> 2;                 // 4 chunks of 8 halves per row
      int kc  = (c & 3) * 8;
      ASYNC_B128(A  + (size_t)(m0 + row) * lda + kt * BK + kc, &As[buf][row][kc]);
      ASYNC_B128(Bt + (size_t)(n0 + row) * ldb + kt * BK + kc, &Bs[buf][row][kc]);
    }
  };
  issue_copy(0, 0);
  for (int kt = 0; kt < nkt; ++kt) {
    const int buf = kt & 1;
    WAIT_ASYNC0();                      // tile kt landed in LDS (this wave's ops)
    __syncthreads();                    // visible to all waves; prev compute done
    if (kt + 1 < nkt) issue_copy(kt + 1, buf ^ 1);
#else
  // ---------------- register-staged software pipeline (fallback) ----------------
  uint4 ra[2], rb[2];
  auto gload = [&](int kt) {
#pragma unroll
    for (int i = 0; i < 2; ++i) {
      int c   = tid + i * 256;
      int row = c >> 2;
      int kc  = (c & 3) * 8;
      ra[i] = *(const uint4*)(A  + (size_t)(m0 + row) * lda + kt * BK + kc);
      rb[i] = *(const uint4*)(Bt + (size_t)(n0 + row) * ldb + kt * BK + kc);
    }
  };
  gload(0);
  for (int kt = 0; kt < nkt; ++kt) {
    const int buf = kt & 1;
#pragma unroll
    for (int i = 0; i < 2; ++i) {
      int c   = tid + i * 256;
      int row = c >> 2;
      int kc  = (c & 3) * 8;
      *(uint4*)&As[buf][row][kc] = ra[i];
      *(uint4*)&Bs[buf][row][kc] = rb[i];
    }
    __syncthreads();
    if (kt + 1 < nkt) gload(kt + 1);    // overlap next-tile loads with compute
#endif

    // ---- fragment gather per documented WMMA VGPR layouts ----
    union { v16h v; uint4 u[2]; } af[4], bf[2];
#pragma unroll
    for (int mi = 0; mi < 4; ++mi) {
      int r = waveM * 64 + mi * 16 + l15;                  // A: M = lane&15
      af[mi].u[0] = *(const uint4*)&As[buf][r][hi * 8];    // K 0-7  / 8-15
      af[mi].u[1] = *(const uint4*)&As[buf][r][hi * 8 + 16]; // K 16-23 / 24-31
    }
#pragma unroll
    for (int ni = 0; ni < 2; ++ni) {
      int r = waveN * 32 + ni * 16 + l15;                  // B: N = lane&15
      bf[ni].u[0] = *(const uint4*)&Bs[buf][r][hi * 8];    // K 0-7  / 8-15
      bf[ni].u[1] = *(const uint4*)&Bs[buf][r][hi * 8 + 16]; // K 16-23 / 24-31
    }
#pragma unroll
    for (int mi = 0; mi < 4; ++mi)
#pragma unroll
      for (int ni = 0; ni < 2; ++ni)
        acc[mi][ni] = __builtin_amdgcn_wmma_f32_16x16x32_f16(
            false, af[mi].v, false, bf[ni].v, (short)0, acc[mi][ni], false, false);
  }

  // ---- epilogue: bias (+relu) (*att) -> f16 or f32 ----
#pragma unroll
  for (int mi = 0; mi < 4; ++mi) {
#pragma unroll
    for (int ni = 0; ni < 2; ++ni) {
      int ng = n0 + waveN * 32 + ni * 16 + l15;
      float bv = (ng < Nvalid) ? bias[ng] : 0.0f;
#pragma unroll
      for (int i = 0; i < 8; ++i) {
        int mg = m0 + waveM * 64 + mi * 16 + hi * 8 + i;   // C: M = vgpr + 8*hi
        float v = acc[mi][ni][i] + bv;
        if (RELU) v = v > 0.0f ? v : 0.0f;
        if (SCALE_ATT) v *= att[mg];
        if (ng < Nvalid) {
          if (OUT_F16) ((_Float16*)Cout)[(size_t)mg * ldc + ng] = (_Float16)v;
          else         ((float*)Cout)[(size_t)mg * ldc + ng] = v;
        }
      }
    }
  }
}

// ==================================================================
// att[m] = sigmoid( sum_k c1[m][k<256] * aw2[k] + ab2 )   (wave per row)
// ==================================================================
__global__ __launch_bounds__(256) void k_att(const _Float16* __restrict__ c1,
                                             const float* __restrict__ aw2,
                                             const float* __restrict__ ab2,
                                             float* __restrict__ att) {
  int m = blockIdx.x * 8 + (threadIdx.x >> 5);
  int lane = threadIdx.x & 31;
  if (m >= Mc) return;
  const _Float16* row = c1 + (size_t)m * N1c;
  float s = 0.0f;
#pragma unroll
  for (int k = lane; k < AHc; k += 32) s += (float)row[k] * aw2[k];
#pragma unroll
  for (int off = 16; off > 0; off >>= 1) s += __shfl_xor(s, off, 32);
  if (lane == 0) att[m] = 1.0f / (1.0f + __expf(-(s + ab2[0])));
}

// ==================================================================
// outH[l,b,h] = cumsum_l encAtt[l,b,h]   (f32 accum over 6 steps)
// ==================================================================
__global__ __launch_bounds__(256) void k_cumsum(const _Float16* __restrict__ encA,
                                                _Float16* __restrict__ outH) {
  int idx = blockIdx.x * 256 + threadIdx.x;   // b*H + h, total 1<<20
  int b = idx >> 10, h = idx & 1023;
  float acc = 0.0f;
#pragma unroll
  for (int l = 0; l < Lc; ++l) {
    size_t o = ((size_t)(l * Bc + b)) * Hc + h;
    acc += (float)encA[o];
    outH[o] = (_Float16)acc;
  }
}

// ==================================================================
// host launcher
// ==================================================================
extern "C" void kernel_launch(void* const* d_in, const int* in_sizes, int n_in,
                              void* d_out, int out_size, void* d_ws, size_t ws_size,
                              hipStream_t stream) {
  (void)in_sizes; (void)n_in; (void)out_size; (void)ws_size;
  const float* x   = (const float*)d_in[0];
  const float* aw1 = (const float*)d_in[1];
  const float* ab1 = (const float*)d_in[2];
  const float* aw2 = (const float*)d_in[3];
  const float* ab2 = (const float*)d_in[4];
  const float* ew1 = (const float*)d_in[5];
  const float* eb1 = (const float*)d_in[6];
  const float* ew2 = (const float*)d_in[7];
  const float* eb2 = (const float*)d_in[8];
  const float* qw1 = (const float*)d_in[9];
  const float* qb1 = (const float*)d_in[10];
  const float* qw2 = (const float*)d_in[11];
  const float* qb2 = (const float*)d_in[12];
  const float* qw3 = (const float*)d_in[13];
  const float* qb3 = (const float*)d_in[14];
  float* out = (float*)d_out;

  char* wsp = (char*)d_ws;
  auto alloc = [&](size_t bytes) -> char* {
    char* p = wsp;
    wsp += (bytes + 255) & ~(size_t)255;
    return p;
  };
  _Float16* inpH = (_Float16*)alloc((size_t)Mc * K1c * 2);
  _Float16* w1T  = (_Float16*)alloc((size_t)N1c * K1c * 2);
  float*    b1c  = (float*)alloc((size_t)N1c * 4);
  _Float16* ew2T = (_Float16*)alloc((size_t)Hc * Hc * 2);
  _Float16* qw1T = (_Float16*)alloc((size_t)Hc * Hc * 2);
  _Float16* qw2T = (_Float16*)alloc((size_t)Hc * Hc * 2);
  _Float16* qw3T = (_Float16*)alloc((size_t)ACTP * Hc * 2);
  _Float16* c1   = (_Float16*)alloc((size_t)Mc * N1c * 2);
  float*    att  = (float*)alloc((size_t)Mc * 4);
  _Float16* encA = (_Float16*)alloc((size_t)Mc * Hc * 2);
  _Float16* outH = (_Float16*)alloc((size_t)Mc * Hc * 2);
  _Float16* h1   = (_Float16*)alloc((size_t)Mc * Hc * 2);
  _Float16* h2   = (_Float16*)alloc((size_t)Mc * Hc * 2);

  auto blocks = [](size_t n) { return (unsigned)((n + 255) / 256); };

  // ---- prep ----
  k_pack_inp<<<blocks((size_t)Mc * K1c), 256, 0, stream>>>(x, inpH);
  // w1T rows [0,256) <- aw1^T, rows [256,1280) <- ew1^T (K padded 518->544)
  {
    dim3 ga(K1c / 32, AHc / 32);
    k_transpose_t<<<ga, 256, 0, stream>>>(aw1, w1T, Nc + Gc, K1c, AHc, AHc);
    dim3 ge(K1c / 32, Hc / 32);
    k_transpose_t<<<ge, 256, 0, stream>>>(ew1, w1T + (size_t)AHc * K1c,
                                          Nc + Gc, K1c, Hc, Hc);
  }
  k_biascat<<<blocks(N1c), 256, 0, stream>>>(ab1, eb1, b1c);
  {
    dim3 g(Hc / 32, Hc / 32);
    k_transpose_t<<<g, 256, 0, stream>>>(ew2, ew2T, Hc, Hc, Hc, Hc);
    k_transpose_t<<<g, 256, 0, stream>>>(qw1, qw1T, Hc, Hc, Hc, Hc);
    k_transpose_t<<<g, 256, 0, stream>>>(qw2, qw2T, Hc, Hc, Hc, Hc);
    dim3 g3(Hc / 32, ACTP / 32);
    k_transpose_t<<<g3, 256, 0, stream>>>(qw3, qw3T, Hc, Hc, ACTc, ACTP);
  }

  // ---- G1: c1 = relu(inp @ [aw1|ew1] + [ab1|eb1])  [M x 1280] f16 ----
  {
    dim3 g(N1c / 128, Mc / 128);
    k_gemm<true, false, true><<<g, 256, 0, stream>>>(
        inpH, K1c, w1T, K1c, b1c, nullptr, c1, N1c, N1c, K1c);
  }
  // ---- att = sigmoid(a1 @ aw2 + ab2) ----
  k_att<<<Mc / 8, 256, 0, stream>>>(c1, aw2, ab2, att);

  // ---- G2: encAtt = att * (e1 @ ew2 + eb2)  [M x 1024] f16 ----
  {
    dim3 g(Hc / 128, Mc / 128);
    k_gemm<false, true, true><<<g, 256, 0, stream>>>(
        c1 + AHc, N1c, ew2T, Hc, eb2, att, encA, Hc, Hc, Hc);
  }
  // ---- cumsum over L ----
  k_cumsum<<<blocks((size_t)Bc * Hc), 256, 0, stream>>>(encA, outH);

  // ---- G3: h1 = relu(out @ qw1 + qb1) ----
  {
    dim3 g(Hc / 128, Mc / 128);
    k_gemm<true, false, true><<<g, 256, 0, stream>>>(
        outH, Hc, qw1T, Hc, qb1, nullptr, h1, Hc, Hc, Hc);
  }
  // ---- G4: h2 = relu(h1 @ qw2 + qb2) ----
  {
    dim3 g(Hc / 128, Mc / 128);
    k_gemm<true, false, true><<<g, 256, 0, stream>>>(
        h1, Hc, qw2T, Hc, qb2, nullptr, h2, Hc, Hc, Hc);
  }
  // ---- G5: action = h2 @ qw3 + qb3  [M x 12972] f32 ----
  {
    dim3 g(ACTP / 128, Mc / 128);
    k_gemm<false, false, false><<<g, 256, 0, stream>>>(
        h2, Hc, qw3T, Hc, qb3, nullptr, out, ACTc, ACTc, Hc);
  }
}
